// GraphTransformerModel_83614423318801
// MI455X (gfx1250) — matrix operand order repaired and checked by reading, not compiled, not run on hardware
//
#include <hip/hip_runtime.h>
#include <hip/hip_bf16.h>

typedef __attribute__((ext_vector_type(16))) _Float16 v16h;
typedef __attribute__((ext_vector_type(8)))  _Float16 v8h;
typedef __attribute__((ext_vector_type(8)))  float    v8f;
typedef int v4i __attribute__((__vector_size__(16)));

#define NB 256      // nodes per batch
#define DMODEL 128
#define NHEAD 8
#define DHEAD 16
#define DEDGE 8
#define NBATCH 4
#define NROWS (NBATCH * NB)        // 1024
#define NBH (NBATCH * NHEAD)       // 32

#if __has_builtin(__builtin_amdgcn_global_load_async_to_lds_b128)
#define HAVE_ASYNC_LDS 1
typedef __attribute__((address_space(1))) v4i* gptr_v4i;
typedef __attribute__((address_space(3))) v4i* lptr_v4i;
#endif

__device__ __forceinline__ void wait_async_zero() {
#if __has_builtin(__builtin_amdgcn_s_wait_asynccnt)
    __builtin_amdgcn_s_wait_asynccnt(0);
#else
    asm volatile("s_wait_asynccnt 0x0" ::: "memory");
#endif
}

// ---------------------------------------------------------------------------
// Projection GEMM: X(1024x128) x W(128x128) (+bias) for Wq/Wk/Wv/Wres.
// One wave per 16x16 output tile, K-loop of 4x32 via v_wmma_f32_16x16x32_f16.
// B staged TRANSPOSED in LDS so fragments load as ds_load_b128 pairs.
// ---------------------------------------------------------------------------
__global__ __launch_bounds__(32)
void proj_wmma_kernel(const float* __restrict__ X,
                      const float* __restrict__ Wq, const float* __restrict__ bq,
                      const float* __restrict__ Wk, const float* __restrict__ bk,
                      const float* __restrict__ Wv, const float* __restrict__ bv,
                      const float* __restrict__ Wr, const float* __restrict__ br,
                      _Float16* __restrict__ qo, _Float16* __restrict__ ko,
                      _Float16* __restrict__ vo, float* __restrict__ ro)
{
    const int mat = blockIdx.z;        // 0=q 1=k 2=v 3=res
    const int n0  = blockIdx.y * 16;
    const int m0  = blockIdx.x * 16;
    const int t   = threadIdx.x;       // 0..31 (wave32)

    const float* W; const float* bias;
    switch (mat) {
        case 0:  W = Wq; bias = bq; break;
        case 1:  W = Wk; bias = bk; break;
        case 2:  W = Wv; bias = bv; break;
        default: W = Wr; bias = br; break;
    }

    __shared__ __align__(16) _Float16 Asub[16 * 32];   // [row][k]
    __shared__ __align__(16) _Float16 BsubT[16 * 32];  // [col][k]  (transposed!)

    v8f acc = {};
    for (int k0 = 0; k0 < DMODEL; k0 += 32) {
        for (int idx = t; idx < 512; idx += 32) {
            const int ra = idx >> 5, ca = idx & 31;              // row, k
            Asub[idx]  = (_Float16)X[(m0 + ra) * DMODEL + (k0 + ca)];
            BsubT[idx] = (_Float16)W[(k0 + ca) * DMODEL + (n0 + ra)];
        }
        __syncthreads();

        // A fragment: lane<16 -> row=lane, K {0..7,16..23}; lane>=16 -> row-16, K {8..15,24..31}
        const int rowA = (t < 16) ? t : (t - 16);
        const int kA   = (t < 16) ? 0 : 8;
        const v8h a0 = *(const v8h*)&Asub[rowA * 32 + kA];
        const v8h a1 = *(const v8h*)&Asub[rowA * 32 + kA + 16];
        // B fragment: lane<16 -> col=lane, K {0..7,16..23}; lane>=16 -> col=lane-16, K {8..15,24..31}
        const int colB = t & 15;
        const int kB   = (t < 16) ? 0 : 8;
        const v8h b0 = *(const v8h*)&BsubT[colB * 32 + kB];
        const v8h b1 = *(const v8h*)&BsubT[colB * 32 + kB + 16];

        v16h a, bf;
        #pragma unroll
        for (int x = 0; x < 8; ++x) {
            a[x] = a0[x]; a[x + 8] = a1[x];
            bf[x] = b0[x]; bf[x + 8] = b1[x];
        }
        acc = __builtin_amdgcn_wmma_f32_16x16x32_f16(false, a, false, bf,
                                                     (short)0, acc, false, false);
        __syncthreads();
    }

    const int n  = t & 15;
    const int mb = (t < 16) ? 0 : 8;
    #pragma unroll
    for (int r = 0; r < 8; ++r) {
        const int m = m0 + mb + r;
        const int j = n0 + n;
        const float val = acc[r] + bias[j];
        if (mat == 3) {
            ro[m * DMODEL + j] = val;
        } else {
            const int b = m >> 8, i = m & 255, h = j >> 4, d = j & 15;
            _Float16* dst = (mat == 0) ? qo : (mat == 1) ? ko : vo;
            dst[(((b * NHEAD + h) * NB) + i) * DHEAD + d] = (_Float16)val;
        }
    }
}

// ---------------------------------------------------------------------------
// u[row,c] = We_h^T q_row  (8-vec per (b,h,i) row), qbe[row] = q_row . be_h
// ---------------------------------------------------------------------------
__global__ __launch_bounds__(256)
void u_kernel(const _Float16* __restrict__ q, const float* __restrict__ We,
              const float* __restrict__ be, float* __restrict__ u,
              float* __restrict__ qbe)
{
    const int row = blockIdx.x * blockDim.x + threadIdx.x;   // (b*8+h)*256+i
    if (row >= NBH * NB) return;
    const int h = (row >> 8) & 7;
    const _Float16* qr = q + row * DHEAD;
    float qv[DHEAD];
    #pragma unroll
    for (int d = 0; d < DHEAD; ++d) qv[d] = (float)qr[d];
    #pragma unroll
    for (int c = 0; c < DEDGE; ++c) {
        float s = 0.f;
        #pragma unroll
        for (int d = 0; d < DHEAD; ++d) s += qv[d] * We[c * DMODEL + h * DHEAD + d];
        u[row * DEDGE + c] = s;
    }
    float s = 0.f;
    #pragma unroll
    for (int d = 0; d < DHEAD; ++d) s += qv[d] * be[h * DHEAD + d];
    qbe[row] = s;
}

// ---------------------------------------------------------------------------
// means[i] = mean_j sim[bh=0, i, j]  (the reference's quirky [0]-indexed shift)
// ---------------------------------------------------------------------------
__global__ __launch_bounds__(256)
void means_kernel(const _Float16* __restrict__ q, const _Float16* __restrict__ k,
                  const float* __restrict__ edges, const float* __restrict__ u,
                  const float* __restrict__ qbe, float* __restrict__ means,
                  float scale)
{
    const int i = blockIdx.x;
    const int j = threadIdx.x;
    const _Float16* qr = q + i * DHEAD;          // bh = 0
    const _Float16* kr = k + j * DHEAD;
    float dot = 0.f;
    #pragma unroll
    for (int d = 0; d < DHEAD; ++d) dot += (float)qr[d] * (float)kr[d];
    const float* er = edges + (i * NB + j) * DEDGE;  // b = 0
    const float* ur = u + i * DEDGE;
    float ed = 0.f;
    #pragma unroll
    for (int c = 0; c < DEDGE; ++c) ed += er[c] * ur[c];
    const float sim = (dot + ed + qbe[i]) * scale;

    __shared__ float red[256];
    red[j] = sim; __syncthreads();
    for (int s = 128; s > 0; s >>= 1) {
        if (j < s) red[j] += red[j + s];
        __syncthreads();
    }
    if (j == 0) means[i] = red[0] / (float)NB;
}

// ---------------------------------------------------------------------------
// Fused attention for one (bh, 16-row i-tile).
//   V tile staged via GLOBAL_LOAD_ASYNC_TO_LDS_B128 (ASYNCcnt), transposed
//   once in LDS; single edge pass accumulates probs + edge moments; 8 waves
//   each run one K=32 chunk of P(16x256) x V(256x16) WMMA with ds_add_f32
//   accumulate; the 1/rowsum normalization is folded into the epilogue.
// ---------------------------------------------------------------------------
__global__ __launch_bounds__(256)
void attn_kernel(const _Float16* __restrict__ q, const _Float16* __restrict__ k,
                 const _Float16* __restrict__ v, const float* __restrict__ edges,
                 const float* __restrict__ adj, const float* __restrict__ u,
                 const float* __restrict__ qbe, const float* __restrict__ means,
                 const float* __restrict__ We, const float* __restrict__ be,
                 float* __restrict__ outb, float scale)
{
    const int bh = blockIdx.y; const int b = bh >> 3; const int h = bh & 7;
    const int i0 = blockIdx.x * 16;
    const int t  = threadIdx.x;
    const int ip = t >> 4;     // i' in tile (0..15)
    const int jl = t & 15;     // j stripe (0..15)

    __shared__ __align__(16) _Float16 attnH[16 * NB];   // 8 KB unnormalized probs
    __shared__ __align__(16) _Float16 vtRM[NB * DHEAD]; // 8 KB V rows (async staged)
    __shared__ __align__(16) _Float16 vtT[DHEAD * NB];  // 8 KB V transposed
    __shared__ float AV[16 * 16];
    __shared__ float psum[256];
    __shared__ float Sinv[16];
    __shared__ float wpart[16 * 16 * DEDGE];
    __shared__ float spart[256];
    __shared__ float wrow[16 * DEDGE];
    __shared__ float srow2[16];

    // --- stage V (256 x 16 f16): thread t owns row j = t --------------------
    {
        const _Float16* vr = v + (bh * NB + t) * DHEAD;
#ifdef HAVE_ASYNC_LDS
        gptr_v4i gp0 = (gptr_v4i)(v4i*)(void*)vr;
        gptr_v4i gp1 = (gptr_v4i)(v4i*)(void*)(vr + 8);
        lptr_v4i lp0 = (lptr_v4i)(v4i*)(void*)&vtRM[t * DHEAD];
        lptr_v4i lp1 = (lptr_v4i)(v4i*)(void*)&vtRM[t * DHEAD + 8];
        __builtin_amdgcn_global_load_async_to_lds_b128(gp0, lp0, 0, 0);
        __builtin_amdgcn_global_load_async_to_lds_b128(gp1, lp1, 0, 0);
        wait_async_zero();
#else
        #pragma unroll
        for (int d = 0; d < DHEAD; ++d) vtRM[t * DHEAD + d] = vr[d];
#endif
    }
    AV[t & 255] = 0.f;
    // transpose own row into vtT (own-wave async already waited; row t is ours)
    {
        const v8h r0 = *(const v8h*)&vtRM[t * DHEAD];
        const v8h r1 = *(const v8h*)&vtRM[t * DHEAD + 8];
        #pragma unroll
        for (int d = 0; d < 8; ++d) {
            vtT[d * NB + t]       = r0[d];
            vtT[(d + 8) * NB + t] = r1[d];
        }
    }

    const int i    = i0 + ip;
    const int qrow = bh * NB + i;
    float qv[DHEAD];
    {
        const _Float16* qr = q + qrow * DHEAD;
        #pragma unroll
        for (int d = 0; d < DHEAD; ++d) qv[d] = (float)qr[d];
    }
    float uv[DEDGE];
    {
        const float* ur = u + qrow * DEDGE;
        #pragma unroll
        for (int c = 0; c < DEDGE; ++c) uv[c] = ur[c];
    }
    const float qb = qbe[qrow];
    const float mi = means[i];

    // --- phase 1: sim -> masked exp (raw) + edge moments in the same pass ---
    float ps = 0.f;
    float wc[DEDGE] = {0.f, 0.f, 0.f, 0.f, 0.f, 0.f, 0.f, 0.f};
    for (int j = jl; j < NB; j += 16) {
        const _Float16* kr = k + (bh * NB + j) * DHEAD;
        float dot = 0.f;
        #pragma unroll
        for (int d = 0; d < DHEAD; ++d) dot += qv[d] * (float)kr[d];
        const float* er = edges + ((size_t)(b * NB + i) * NB + j) * DEDGE;
        __builtin_prefetch(er + 16 * DEDGE, 0, 0);   // global_prefetch_b8
        float ed = 0.f;
        #pragma unroll
        for (int c = 0; c < DEDGE; ++c) ed += er[c] * uv[c];
        const float sim = (dot + ed + qb) * scale;
        const float p = __expf(sim - mi) * adj[(size_t)(b * NB + i) * NB + j];
        attnH[ip * NB + j] = (_Float16)p;            // unnormalized
        ps += p;
        #pragma unroll
        for (int c = 0; c < DEDGE; ++c) wc[c] += p * er[c];
    }
    psum[t] = ps;
    spart[t] = ps;
    #pragma unroll
    for (int c = 0; c < DEDGE; ++c) wpart[(ip * 16 + jl) * DEDGE + c] = wc[c];
    __syncthreads();

    if (jl == 0) {
        float s = 0.f;
        #pragma unroll
        for (int x = 0; x < 16; ++x) s += psum[ip * 16 + x];
        const float si = (s == 0.f) ? 1.f : (1.f / s);
        Sinv[ip] = si;
        srow2[ip] = s * si;          // 1 (or 0 for fully-masked rows)
    }
    if (t < 128) {                   // raw w reduction over the 16 j-stripes
        const int rp = t >> 3, c = t & 7;
        float s = 0.f;
        #pragma unroll
        for (int x = 0; x < 16; ++x) s += wpart[(rp * 16 + x) * DEDGE + c];
        wrow[rp * DEDGE + c] = s;
    }
    __syncthreads();

    // --- phase 2: P(16x256) x V(256x16); wave w handles K chunk [32w,32w+32)
    {
        const int wave = t >> 5, lane = t & 31;
        const int kc = wave * 32;
        const int rowA = (lane < 16) ? lane : lane - 16;
        const int colB = lane & 15;
        const int kA = kc + ((lane < 16) ? 0 : 8);
        const int kB = kc + ((lane < 16) ? 0 : 8);
        const v8h a0 = *(const v8h*)&attnH[rowA * NB + kA];
        const v8h a1 = *(const v8h*)&attnH[rowA * NB + kA + 16];
        const v8h b0 = *(const v8h*)&vtT[colB * NB + kB];
        const v8h b1 = *(const v8h*)&vtT[colB * NB + kB + 16];
        v16h a, bf;
        #pragma unroll
        for (int x = 0; x < 8; ++x) {
            a[x] = a0[x]; a[x + 8] = a1[x];
            bf[x] = b0[x]; bf[x + 8] = b1[x];
        }
        v8f acc = {};
        acc = __builtin_amdgcn_wmma_f32_16x16x32_f16(false, a, false, bf,
                                                     (short)0, acc, false, false);
        const int mb = (lane < 16) ? 0 : 8;
        #pragma unroll
        for (int r = 0; r < 8; ++r)
            atomicAdd(&AV[(mb + r) * 16 + colB], acc[r]);   // ds_add_f32
    }
    __syncthreads();

    // --- epilogue: out = Sinv * (AVraw + w_raw . We) + s_norm * be ----------
    {
        const int d = jl;
        const int col = h * DHEAD + d;
        float o = AV[ip * 16 + d];
        #pragma unroll
        for (int c = 0; c < DEDGE; ++c) o += wrow[ip * DEDGE + c] * We[c * DMODEL + col];
        o = o * Sinv[ip] + srow2[ip] * be[col];
        outb[(size_t)(b * NB + i0 + ip) * DMODEL + col] = o;
    }
}

// ---------------------------------------------------------------------------
// Gated residual + LayerNorm + ReLU (one 128-wide row per block)
// ---------------------------------------------------------------------------
__global__ __launch_bounds__(128)
void gate_ln_kernel(const float* __restrict__ outb, const float* __restrict__ res,
                    const float* __restrict__ Wg, const float* __restrict__ lng,
                    const float* __restrict__ lnb, float* __restrict__ xout)
{
    const int row = blockIdx.x;
    const int d = threadIdx.x;
    const float o = outb[row * DMODEL + d];
    const float r = res[row * DMODEL + d];
    __shared__ float red[128];

    red[d] = o * Wg[d] + r * Wg[DMODEL + d] + (o - r) * Wg[2 * DMODEL + d];
    __syncthreads();
    for (int s = 64; s > 0; s >>= 1) { if (d < s) red[d] += red[d + s]; __syncthreads(); }
    const float g = 1.f / (1.f + __expf(-red[0]));
    __syncthreads();

    const float y = o * g + r * (1.f - g);
    red[d] = y; __syncthreads();
    for (int s = 64; s > 0; s >>= 1) { if (d < s) red[d] += red[d + s]; __syncthreads(); }
    const float mu = red[0] * (1.f / DMODEL);
    __syncthreads();

    const float yc = y - mu;
    red[d] = yc * yc; __syncthreads();
    for (int s = 64; s > 0; s >>= 1) { if (d < s) red[d] += red[d + s]; __syncthreads(); }
    const float var = red[0] * (1.f / DMODEL);
    __syncthreads();

    const float yn = yc * __frsqrt_rn(var + 1e-5f) * lng[d] + lnb[d];
    xout[row * DMODEL + d] = fmaxf(yn, 0.f);
}

// ---------------------------------------------------------------------------
extern "C" void kernel_launch(void* const* d_in, const int* in_sizes, int n_in,
                              void* d_out, int out_size, void* d_ws, size_t ws_size,
                              hipStream_t stream)
{
    const float* nodes = (const float*)d_in[0];
    const float* edges = (const float*)d_in[1];
    const float* adjacency = (const float*)d_in[2];
    const float* Wq = (const float*)d_in[3];
    const float* bq = (const float*)d_in[4];
    const float* Wk = (const float*)d_in[5];
    const float* bk = (const float*)d_in[6];
    const float* Wv = (const float*)d_in[7];
    const float* bv = (const float*)d_in[8];
    const float* We = (const float*)d_in[9];
    const float* be = (const float*)d_in[10];
    const float* Wres = (const float*)d_in[11];
    const float* bres = (const float*)d_in[12];
    const float* Wgate = (const float*)d_in[13];
    const float* lng = (const float*)d_in[14];
    const float* lnb = (const float*)d_in[15];
    float* out = (float*)d_out;

    char* ws = (char*)d_ws;
    _Float16* qb   = (_Float16*)(ws + 0);            // 256 KB
    _Float16* kb   = (_Float16*)(ws + (256 << 10));
    _Float16* vb   = (_Float16*)(ws + (512 << 10));
    float*    resb = (float*)   (ws + (768 << 10));  // 512 KB
    float*    outw = (float*)   (ws + (1280 << 10));
    float*    ub   = (float*)   (ws + (1792 << 10)); // 256 KB
    float*    qbeb = (float*)   (ws + (2048 << 10)); // 32 KB
    float*    meansb = (float*) (ws + (2080 << 10)); // 1 KB
    float*    x1   = (float*)   (ws + (2084 << 10)); // 512 KB

    const float scale = 0.25f;   // 1/sqrt(16)
    const float* xin = nodes;
    for (int l = 0; l < 2; ++l) {
        float* xout = (l == 1) ? out : x1;
        proj_wmma_kernel<<<dim3(NROWS / 16, DMODEL / 16, 4), 32, 0, stream>>>(
            xin,
            Wq + l * DMODEL * DMODEL, bq + l * DMODEL,
            Wk + l * DMODEL * DMODEL, bk + l * DMODEL,
            Wv + l * DMODEL * DMODEL, bv + l * DMODEL,
            Wres + l * DMODEL * DMODEL, bres + l * DMODEL,
            qb, kb, vb, resb);
        u_kernel<<<NBH * NB / 256, 256, 0, stream>>>(
            qb, We + l * DEDGE * DMODEL, be + l * DMODEL, ub, qbeb);
        means_kernel<<<NB, NB, 0, stream>>>(qb, kb, edges, ub, qbeb, meansb, scale);
        attn_kernel<<<dim3(NB / 16, NBH), 256, 0, stream>>>(
            qb, kb, vb, edges, adjacency, ub, qbeb, meansb,
            We + l * DEDGE * DMODEL, be + l * DMODEL, outw, scale);
        gate_ln_kernel<<<NROWS, DMODEL, 0, stream>>>(
            outw, resb, Wgate + l * 3 * DMODEL, lng + l * DMODEL, lnb + l * DMODEL, xout);
        xin = xout;
    }
}
